// EdgeNet_35244501631590
// MI455X (gfx1250) — matrix operand order repaired and checked by reading, not compiled, not run on hardware
//
#include <hip/hip_runtime.h>

typedef __attribute__((ext_vector_type(16))) _Float16 v16h;
typedef __attribute__((ext_vector_type(8)))  _Float16 v8h;
typedef __attribute__((ext_vector_type(8)))  float    v8f;

#define EPS 1e-5f
#define WAVES_PER_BLOCK 4

// ---------------------------------------------------------------------------
// Zero a float region
// ---------------------------------------------------------------------------
__global__ void zero_kernel(float* __restrict__ p, long long n) {
    long long i = (long long)blockIdx.x * blockDim.x + threadIdx.x;
    long long stride = (long long)gridDim.x * blockDim.x;
    for (; i < n; i += stride) p[i] = 0.0f;
}

// ---------------------------------------------------------------------------
// BatchNorm stats for x [N,3]: stats[0..2]=sum, stats[3..5]=sumsq
// ---------------------------------------------------------------------------
__global__ void bn_stats_x_kernel(const float* __restrict__ x, float* __restrict__ stats,
                                  long long N) {
    __shared__ float sh[6][256];
    float s[6] = {0.f, 0.f, 0.f, 0.f, 0.f, 0.f};
    long long i = (long long)blockIdx.x * blockDim.x + threadIdx.x;
    long long stride = (long long)gridDim.x * blockDim.x;
    for (; i < N; i += stride) {
        float a0 = x[i * 3 + 0], a1 = x[i * 3 + 1], a2 = x[i * 3 + 2];
        s[0] += a0; s[1] += a1; s[2] += a2;
        s[3] += a0 * a0; s[4] += a1 * a1; s[5] += a2 * a2;
    }
    for (int j = 0; j < 6; ++j) sh[j][threadIdx.x] = s[j];
    __syncthreads();
    for (int off = 128; off > 0; off >>= 1) {
        if ((int)threadIdx.x < off)
            for (int j = 0; j < 6; ++j) sh[j][threadIdx.x] += sh[j][threadIdx.x + off];
        __syncthreads();
    }
    if (threadIdx.x == 0)
        for (int j = 0; j < 6; ++j) atomicAdd(&stats[j], sh[j][0]);
}

// ---------------------------------------------------------------------------
// Apply BN to x, write padded float4 per node: xn4[n] = {xn0, xn1, xn2, 0}
// ---------------------------------------------------------------------------
__global__ void bn_apply_x_kernel(const float* __restrict__ x, const float* __restrict__ stats,
                                  const float* __restrict__ gamma, const float* __restrict__ beta,
                                  float* __restrict__ xn4, long long N) {
    float invN = 1.0f / (float)N;
    float m[3], sc[3], bb[3];
    for (int c = 0; c < 3; ++c) {
        m[c] = stats[c] * invN;
        float v = stats[3 + c] * invN - m[c] * m[c];
        sc[c] = rsqrtf(v + EPS) * gamma[c];
        bb[c] = beta[c];
    }
    long long i = (long long)blockIdx.x * blockDim.x + threadIdx.x;
    long long stride = (long long)gridDim.x * blockDim.x;
    for (; i < N; i += stride) {
        xn4[i * 4 + 0] = (x[i * 3 + 0] - m[0]) * sc[0] + bb[0];
        xn4[i * 4 + 1] = (x[i * 3 + 1] - m[1]) * sc[1] + bb[1];
        xn4[i * 4 + 2] = (x[i * 3 + 2] - m[2]) * sc[2] + bb[2];
        xn4[i * 4 + 3] = 0.0f;
    }
}

// ---------------------------------------------------------------------------
// BN for u [G,2] (single block)
// ---------------------------------------------------------------------------
__global__ void bn_u_kernel(const float* __restrict__ u, const float* __restrict__ gamma,
                            const float* __restrict__ beta, float* __restrict__ u1, int G) {
    __shared__ float sh[4][256];
    float s0 = 0.f, s1 = 0.f, q0 = 0.f, q1 = 0.f;
    for (int i = threadIdx.x; i < G; i += 256) {
        float a = u[i * 2 + 0], b = u[i * 2 + 1];
        s0 += a; s1 += b; q0 += a * a; q1 += b * b;
    }
    sh[0][threadIdx.x] = s0; sh[1][threadIdx.x] = s1;
    sh[2][threadIdx.x] = q0; sh[3][threadIdx.x] = q1;
    __syncthreads();
    for (int off = 128; off > 0; off >>= 1) {
        if ((int)threadIdx.x < off)
            for (int j = 0; j < 4; ++j) sh[j][threadIdx.x] += sh[j][threadIdx.x + off];
        __syncthreads();
    }
    float invG = 1.0f / (float)G;
    float m0 = sh[0][0] * invG, m1 = sh[1][0] * invG;
    float v0 = sh[2][0] * invG - m0 * m0, v1 = sh[3][0] * invG - m1 * m1;
    float sc0 = rsqrtf(v0 + EPS) * gamma[0], sc1 = rsqrtf(v1 + EPS) * gamma[1];
    for (int i = threadIdx.x; i < G; i += 256) {
        u1[i * 2 + 0] = (u[i * 2 + 0] - m0) * sc0 + beta[0];
        u1[i * 2 + 1] = (u[i * 2 + 1] - m1) * sc1 + beta[1];
    }
}

// ---------------------------------------------------------------------------
// WMMA helpers
// ---------------------------------------------------------------------------
// B fragment for a 32x16 (KxN) tile of W[fan_in, fan_out] (row-major), output
// columns [n0, n0+16). Layout per ISA 16-bit B tables: lanes 0-15 hold K={0..7, 16..23}
// for column n=lane; lanes 16-31 hold K={8..15, 24..31} for column n=lane-16.
__device__ __forceinline__ v16h make_b_frag(const float* __restrict__ W, int fan_in,
                                            int fan_out, int n0, int lane) {
    int n = n0 + (lane & 15);
    int koff = (lane < 16) ? 0 : 8;
    v16h b;
    for (int e = 0; e < 16; ++e) {
        int k = koff + (e & 7) + ((e >> 3) << 4);
        float w = (k < fan_in) ? W[k * fan_out + n] : 0.0f;
        b[e] = (_Float16)w;
    }
    return b;
}

// A fragment from a row-major 16x32 half tile in LDS. Per ISA 16-bit A layout:
// lanes 0-15: row=lane, K = {0..7, 16..23}; lanes 16-31: row=lane-16, K={8..15, 24..31}.
__device__ __forceinline__ v16h load_a_frag(const _Float16* __restrict__ buf, int lane) {
    int r = lane & 15;
    int kb = (lane < 16) ? 0 : 8;
    const _Float16* rp = buf + r * 32;
    v8h lo = *(const v8h*)(rp + kb);
    v8h hi = *(const v8h*)(rp + kb + 16);
    v16h a;
    for (int i = 0; i < 8; ++i) { a[i] = lo[i]; a[8 + i] = hi[i]; }
    return a;
}

// ---------------------------------------------------------------------------
// Edge MLP: per 16-edge tile, 3-layer MLP via 6x v_wmma_f32_16x16x32_f16,
// then atomic scatter-add into agg[N,32] / deg[N].
// ---------------------------------------------------------------------------
__global__ void __launch_bounds__(128) edge_mlp_kernel(
    const float* __restrict__ xn4,      // [N,4] BN'd node features (padded)
    const int* __restrict__ ei,         // [2,E] (src row then dst row)
    const float* __restrict__ W1, const float* __restrict__ b1,
    const float* __restrict__ W2, const float* __restrict__ b2,
    const float* __restrict__ W3, const float* __restrict__ b3,
    float* __restrict__ agg,            // [N,32]
    float* __restrict__ deg,            // [N]
    long long E) {
    __shared__ __align__(16) _Float16 sMsg[WAVES_PER_BLOCK][16 * 32];
    __shared__ __align__(16) _Float16 sHid[WAVES_PER_BLOCK][16 * 32];
    __shared__ int sDst[WAVES_PER_BLOCK][16];

    const int lane = threadIdx.x & 31;
    const int w    = threadIdx.x >> 5;
    const int nl   = lane & 15;
    const int rbase = (lane < 16) ? 0 : 8;

    // Per-wave persistent weight fragments (in VGPRs)
    v16h B1t0 = make_b_frag(W1, 6, 32, 0, lane);
    v16h B1t1 = make_b_frag(W1, 6, 32, 16, lane);
    v16h B2t0 = make_b_frag(W2, 32, 32, 0, lane);
    v16h B2t1 = make_b_frag(W2, 32, 32, 16, lane);
    v16h B3t0 = make_b_frag(W3, 32, 32, 0, lane);
    v16h B3t1 = make_b_frag(W3, 32, 32, 16, lane);
    float bias1_0 = b1[nl], bias1_1 = b1[16 + nl];
    float bias2_0 = b2[nl], bias2_1 = b2[16 + nl];
    float bias3_0 = b3[nl], bias3_1 = b3[16 + nl];

    // Zero message tile once: K=6..31 stay zero forever, K=0..5 rewritten per tile.
    for (int i = lane; i < 16 * 32; i += 32) sMsg[w][i] = (_Float16)0.0f;
    asm volatile("s_wait_dscnt 0" ::: "memory");

    const v8f zacc = {};
    const float4* __restrict__ xp = (const float4*)xn4;

    long long nTiles = (E + 15) >> 4;
    long long waveId = (long long)blockIdx.x * WAVES_PER_BLOCK + w;
    long long nWaves = (long long)gridDim.x * WAVES_PER_BLOCK;

    for (long long t = waveId; t < nTiles; t += nWaves) {
        // ---- stage 16 edges' messages into LDS (lanes 0-15, one edge each)
        if (lane < 16) {
            long long e = t * 16 + lane;
            int d = -1;
            float m0 = 0.f, m1 = 0.f, m2 = 0.f, m3 = 0.f, m4 = 0.f, m5 = 0.f;
            if (e < E) {
                int s = ei[e];
                d = ei[E + e];
                float4 xi = xp[d];
                float4 xj = xp[s];
                m0 = xi.x; m1 = xi.y; m2 = xi.z;
                m3 = xj.x - xi.x; m4 = xj.y - xi.y; m5 = xj.z - xi.z;
                atomicAdd(&deg[d], 1.0f);
            }
            _Float16* rp = &sMsg[w][lane * 32];
            rp[0] = (_Float16)m0; rp[1] = (_Float16)m1; rp[2] = (_Float16)m2;
            rp[3] = (_Float16)m3; rp[4] = (_Float16)m4; rp[5] = (_Float16)m5;
            sDst[w][lane] = d;
        }
        asm volatile("s_wait_dscnt 0" ::: "memory");

        // ---- layer 1: [16x32(K=6 used)] x [32x32] -> 16x32
        v16h a = load_a_frag(&sMsg[w][0], lane);
        v8f c0 = __builtin_amdgcn_wmma_f32_16x16x32_f16(false, a, false, B1t0,
                                                        (short)0, zacc, false, false);
        v8f c1 = __builtin_amdgcn_wmma_f32_16x16x32_f16(false, a, false, B1t1,
                                                        (short)0, zacc, false, false);
        for (int g = 0; g < 8; ++g) {
            float v0 = fmaxf(c0[g] + bias1_0, 0.0f);
            float v1 = fmaxf(c1[g] + bias1_1, 0.0f);
            sHid[w][(g + rbase) * 32 + nl]      = (_Float16)v0;
            sHid[w][(g + rbase) * 32 + 16 + nl] = (_Float16)v1;
        }
        asm volatile("s_wait_dscnt 0" ::: "memory");

        // ---- layer 2
        a = load_a_frag(&sHid[w][0], lane);
        c0 = __builtin_amdgcn_wmma_f32_16x16x32_f16(false, a, false, B2t0,
                                                    (short)0, zacc, false, false);
        c1 = __builtin_amdgcn_wmma_f32_16x16x32_f16(false, a, false, B2t1,
                                                    (short)0, zacc, false, false);
        for (int g = 0; g < 8; ++g) {
            float v0 = fmaxf(c0[g] + bias2_0, 0.0f);
            float v1 = fmaxf(c1[g] + bias2_1, 0.0f);
            sHid[w][(g + rbase) * 32 + nl]      = (_Float16)v0;
            sHid[w][(g + rbase) * 32 + 16 + nl] = (_Float16)v1;
        }
        asm volatile("s_wait_dscnt 0" ::: "memory");

        // ---- layer 3 + scatter
        a = load_a_frag(&sHid[w][0], lane);
        c0 = __builtin_amdgcn_wmma_f32_16x16x32_f16(false, a, false, B3t0,
                                                    (short)0, zacc, false, false);
        c1 = __builtin_amdgcn_wmma_f32_16x16x32_f16(false, a, false, B3t1,
                                                    (short)0, zacc, false, false);
        for (int g = 0; g < 8; ++g) {
            int m = g + rbase;
            int d = sDst[w][m];
            if (d >= 0) {
                float v0 = fmaxf(c0[g] + bias3_0, 0.0f);
                float v1 = fmaxf(c1[g] + bias3_1, 0.0f);
                atomicAdd(&agg[(long long)d * 32 + nl], v0);
                atomicAdd(&agg[(long long)d * 32 + 16 + nl], v1);
            }
        }
    }
}

// ---------------------------------------------------------------------------
// Node finalize (mean over edges) + scatter-sum into graphs
// ---------------------------------------------------------------------------
__global__ void pool_kernel(const float* __restrict__ agg, const float* __restrict__ deg,
                            const int* __restrict__ batch, float* __restrict__ gsum,
                            float* __restrict__ gcnt, long long N) {
    long long total = N * 32;
    long long i = (long long)blockIdx.x * blockDim.x + threadIdx.x;
    long long stride = (long long)gridDim.x * blockDim.x;
    for (; i < total; i += stride) {
        long long n = i >> 5;
        int c = (int)(i & 31);
        float inv = 1.0f / fmaxf(deg[n], 1.0f);
        int b = batch[n];
        atomicAdd(&gsum[(long long)b * 32 + c], agg[i] * inv);
        if (c == 0) atomicAdd(&gcnt[b], 1.0f);
    }
}

// ---------------------------------------------------------------------------
// Head MLP: 34 -> 128 -> 128 -> 1, one block per graph
// ---------------------------------------------------------------------------
__global__ void __launch_bounds__(128) final_mlp_kernel(
    const float* __restrict__ u1, const float* __restrict__ gsum,
    const float* __restrict__ gcnt, const float* __restrict__ Wo1,
    const float* __restrict__ bo1, const float* __restrict__ Wo2,
    const float* __restrict__ bo2, const float* __restrict__ Wo3,
    const float* __restrict__ bo3, float* __restrict__ out) {
    __shared__ float in[34];
    __shared__ float h1[128];
    __shared__ float h2[128];
    __shared__ float red[128];
    int g = blockIdx.x;
    int t = threadIdx.x;
    if (t < 34) {
        in[t] = (t < 2) ? u1[g * 2 + t]
                        : gsum[(long long)g * 32 + (t - 2)] / fmaxf(gcnt[g], 1.0f);
    }
    __syncthreads();
    float a = bo1[t];
    for (int k = 0; k < 34; ++k) a += in[k] * Wo1[k * 128 + t];
    h1[t] = fmaxf(a, 0.0f);
    __syncthreads();
    float c = bo2[t];
    for (int k = 0; k < 128; ++k) c += h1[k] * Wo2[k * 128 + t];
    h2[t] = fmaxf(c, 0.0f);
    __syncthreads();
    red[t] = h2[t] * Wo3[t];
    __syncthreads();
    for (int off = 64; off > 0; off >>= 1) {
        if (t < off) red[t] += red[t + off];
        __syncthreads();
    }
    if (t == 0) out[g] = red[0] + bo3[0];
}

// ---------------------------------------------------------------------------
// Launch
// ---------------------------------------------------------------------------
extern "C" void kernel_launch(void* const* d_in, const int* in_sizes, int n_in,
                              void* d_out, int out_size, void* d_ws, size_t ws_size,
                              hipStream_t stream) {
    const float* x     = (const float*)d_in[0];
    const float* u     = (const float*)d_in[1];
    const int*   ei    = (const int*)d_in[2];
    const int*   batch = (const int*)d_in[3];
    const float* bn_g  = (const float*)d_in[4];
    const float* bn_b  = (const float*)d_in[5];
    const float* bng_g = (const float*)d_in[6];
    const float* bng_b = (const float*)d_in[7];
    const float* W1 = (const float*)d_in[8];
    const float* b1 = (const float*)d_in[9];
    const float* W2 = (const float*)d_in[10];
    const float* b2 = (const float*)d_in[11];
    const float* W3 = (const float*)d_in[12];
    const float* b3 = (const float*)d_in[13];
    const float* Wo1 = (const float*)d_in[14];
    const float* bo1 = (const float*)d_in[15];
    const float* Wo2 = (const float*)d_in[16];
    const float* bo2 = (const float*)d_in[17];
    const float* Wo3 = (const float*)d_in[18];
    const float* bo3 = (const float*)d_in[19];

    long long N = in_sizes[0] / 3;
    int       G = in_sizes[1] / 2;
    long long E = in_sizes[2] / 2;

    // Workspace layout (all offsets multiple of 4 floats => 16B aligned)
    auto al4 = [](size_t v) { return (v + 3) & ~(size_t)3; };
    float* ws = (float*)d_ws;
    size_t off = 0;
    float* stats = ws + off; off += 16;
    float* gcnt  = ws + off; off += al4((size_t)G);
    float* gsum  = ws + off; off += al4((size_t)G * 32);
    float* deg   = ws + off; off += al4((size_t)N);
    float* agg   = ws + off; off += al4((size_t)N * 32);
    long long zeroCount = (long long)off;     // zero everything above
    float* xn4 = ws + off; off += al4((size_t)N * 4);
    float* u1  = ws + off; off += al4((size_t)G * 2);

    // 1) zero accumulators + stats
    {
        int blocks = (int)((zeroCount + 255) / 256);
        if (blocks > 2048) blocks = 2048;
        zero_kernel<<<blocks, 256, 0, stream>>>(ws, zeroCount);
    }
    // 2) BN stats for x
    bn_stats_x_kernel<<<256, 256, 0, stream>>>(x, stats, N);
    // 3) BN apply -> xn4
    {
        int blocks = (int)((N + 255) / 256);
        if (blocks > 2048) blocks = 2048;
        bn_apply_x_kernel<<<blocks, 256, 0, stream>>>(x, stats, bn_g, bn_b, xn4, N);
    }
    // 4) BN for u -> u1
    bn_u_kernel<<<1, 256, 0, stream>>>(u, bng_g, bng_b, u1, G);
    // 5) Edge MLP (WMMA) + scatter
    {
        long long tiles = (E + 15) >> 4;
        long long want = (tiles + WAVES_PER_BLOCK - 1) / WAVES_PER_BLOCK;
        int blocks = (want > 2048) ? 2048 : (int)(want ? want : 1);
        edge_mlp_kernel<<<blocks, 128, 0, stream>>>(xn4, ei, W1, b1, W2, b2, W3, b3,
                                                    agg, deg, E);
    }
    // 6) Node mean + graph pooling
    {
        long long total = N * 32;
        int blocks = (int)((total + 255) / 256);
        if (blocks > 4096) blocks = 4096;
        pool_kernel<<<blocks, 256, 0, stream>>>(agg, deg, batch, gsum, gcnt, N);
    }
    // 7) Head MLP
    final_mlp_kernel<<<G, 128, 0, stream>>>(u1, gsum, gcnt, Wo1, bo1, Wo2, bo2,
                                            Wo3, bo3, (float*)d_out);
}
